// GNNAggregationForNegativeModule_6055903887911
// MI455X (gfx1250) — matrix operand order repaired and checked by reading, not compiled, not run on hardware
//
#include <hip/hip_runtime.h>
#include <stdint.h>

#define GN 16384   // graph nodes (N)
#define GD 128     // embedding width (D)
#define GB 8192    // batch (B)
#define GL 3       // layers (L)
#define NPOS (GB/2)
#define SPLITK 8
#define KSLICE (GN / SPLITK)   // 2048

typedef __attribute__((ext_vector_type(16))) __bf16 v16bf;
typedef __attribute__((ext_vector_type(8)))  float  v8f;

union BFrag {
  uint4    u[2];
  v16bf    v;
  uint16_t s[16];
};

__device__ __forceinline__ uint16_t f2bf(float f) {
  uint32_t u = __float_as_uint(f);
  uint32_t r = u + 0x7FFFu + ((u >> 16) & 1u);   // round-to-nearest-even
  return (uint16_t)(r >> 16);
}

// ---------------------------------------------------------------------------
// prep: Et[d][n] = bf16(E[n][d])  (transposed, so WMMA B-frags are contiguous)
//       Wb       = bf16(W)
// ---------------------------------------------------------------------------
__global__ void prep_kernel(const float* __restrict__ E, const float* __restrict__ W,
                            uint16_t* __restrict__ Et, uint16_t* __restrict__ Wb) {
  int idx = blockIdx.x * blockDim.x + threadIdx.x;   // 0 .. GN*GD-1
  int k = idx / GD;            // node
  int n = idx % GD;            // dim
  Et[(size_t)n * GN + k] = f2bf(E[idx]);
  if (idx < GL * GD * GD) Wb[idx] = f2bf(W[idx]);
}

// ---------------------------------------------------------------------------
// partition: order-preserving split of node_indexes by label (jnp.where order),
// plus labels_out = [1]*npos ++ [0]*nneg as float.
// ---------------------------------------------------------------------------
__global__ void partition_kernel(const int* __restrict__ labels,
                                 const int* __restrict__ nodes,
                                 int* __restrict__ pos, int* __restrict__ neg,
                                 float* __restrict__ labels_out) {
  __shared__ int part[1024];
  int tid = threadIdx.x;
  int base = tid * 8;
  int l[8];
  int cnt = 0;
#pragma unroll
  for (int i = 0; i < 8; ++i) { l[i] = labels[base + i]; cnt += l[i]; }
  part[tid] = cnt;
  __syncthreads();
  if (tid == 0) {
    int run = 0;
    for (int i = 0; i < 1024; ++i) { int t = part[i]; part[i] = run; run += t; }
  }
  __syncthreads();
  int p = part[tid];
#pragma unroll
  for (int i = 0; i < 8; ++i) {
    int gi = base + i;
    if (l[i]) { pos[p] = nodes[gi]; ++p; }
    else      { neg[gi - p] = nodes[gi]; }
    labels_out[gi] = (gi < NPOS) ? 1.0f : 0.0f;
  }
}

// ---------------------------------------------------------------------------
// neg_gather: x_bf16[NPOS + j][d] = bf16(E[neg[j]][d])
// ---------------------------------------------------------------------------
__global__ void neg_gather_kernel(const float* __restrict__ E,
                                  const int* __restrict__ neg,
                                  uint16_t* __restrict__ x) {
  int idx = blockIdx.x * blockDim.x + threadIdx.x;   // 0 .. NPOS*GD-1
  int j = idx / GD;
  int d = idx % GD;
  x[(size_t)(NPOS + j) * GD + d] = f2bf(E[(size_t)neg[j] * GD + d]);
}

// ---------------------------------------------------------------------------
// gemm1: pos_emb = A[pos_nodes] @ E   via v_wmma_f32_16x16x32_bf16
// Split-K x8 (2048 waves). A (HBM stream) is software-pipelined one K-step
// ahead; Et B-fragments come from L2/WGP$. f32 partials -> reduce_kernel.
// ---------------------------------------------------------------------------
__global__ void gemm1_kernel(const float* __restrict__ A,
                             const uint16_t* __restrict__ Et,
                             const int* __restrict__ pos,
                             float* __restrict__ partial) {
  int wave = threadIdx.x >> 5;               // 0..3
  int lane = threadIdx.x & 31;
  int tile  = blockIdx.x >> 1;               // 0..255  (NPOS/16 tiles)
  int split = ((blockIdx.x & 1) << 2) + wave;// 0..7
  int row_in = lane & 15;
  int node = pos[tile * 16 + row_in];
  const float* rp = A + (size_t)node * GN;
  int abase = (lane < 16) ? 0 : 8;           // ISA 16-bit A layout (16x32)
  int koff  = (lane < 16) ? 0 : 8;           // ISA 16-bit B layout (32x16)
  int kbeg = split * KSLICE;
  int kend = kbeg + KSLICE;

  v8f acc[8];
#pragma unroll
  for (int j = 0; j < 8; ++j) acc[j] = (v8f){0.f,0.f,0.f,0.f,0.f,0.f,0.f,0.f};

  // ---- prologue: raw A floats for first K-step ----
  float4 a0 = *(const float4*)(rp + kbeg + abase);
  float4 a1 = *(const float4*)(rp + kbeg + abase + 4);
  float4 a2 = *(const float4*)(rp + kbeg + abase + 16);
  float4 a3 = *(const float4*)(rp + kbeg + abase + 20);

  for (int kc = kbeg; kc < kend; kc += 32) {
    // branchless clamped prefetch of the A stream (~5 iterations ahead)
    int pk = (kc + 160 < kend) ? (kc + 160) : kc;
    __builtin_prefetch(rp + pk, 0, 1);

    // ---- pack current A fragment from pipelined floats ----
    BFrag af;
    af.s[ 0]=f2bf(a0.x); af.s[ 1]=f2bf(a0.y); af.s[ 2]=f2bf(a0.z); af.s[ 3]=f2bf(a0.w);
    af.s[ 4]=f2bf(a1.x); af.s[ 5]=f2bf(a1.y); af.s[ 6]=f2bf(a1.z); af.s[ 7]=f2bf(a1.w);
    af.s[ 8]=f2bf(a2.x); af.s[ 9]=f2bf(a2.y); af.s[10]=f2bf(a2.z); af.s[11]=f2bf(a2.w);
    af.s[12]=f2bf(a3.x); af.s[13]=f2bf(a3.y); af.s[14]=f2bf(a3.z); af.s[15]=f2bf(a3.w);

    // ---- issue next iteration's A loads (overlap HBM latency with WMMAs) ----
    int kn = (kc + 32 < kend) ? (kc + 32) : kc;   // harmless reload on tail
    float4 n0 = *(const float4*)(rp + kn + abase);
    float4 n1 = *(const float4*)(rp + kn + abase + 4);
    float4 n2 = *(const float4*)(rp + kn + abase + 16);
    float4 n3 = *(const float4*)(rp + kn + abase + 20);

    // ---- B fragments + 8 WMMAs ----
    BFrag bf[8];
#pragma unroll
    for (int j = 0; j < 8; ++j) {
      const uint16_t* bp = Et + (size_t)(j * 16 + row_in) * GN + kc + koff;
      bf[j].u[0] = *(const uint4*)bp;
      bf[j].u[1] = *(const uint4*)(bp + 16);
    }
#pragma unroll
    for (int j = 0; j < 8; ++j)
      acc[j] = __builtin_amdgcn_wmma_f32_16x16x32_bf16(
                 false, af.v, false, bf[j].v, (short)0, acc[j], false, false);

    a0 = n0; a1 = n1; a2 = n2; a3 = n3;
  }

  // ---- store f32 partial tile (C/D layout: lanes 16+ carry M+8) ----
  int rowadd = (lane < 16) ? 0 : 8;
  float* pout = partial + (size_t)split * NPOS * GD;
#pragma unroll
  for (int r = 0; r < 8; ++r) {
    size_t row = (size_t)tile * 16 + r + rowadd;
#pragma unroll
    for (int j = 0; j < 8; ++j)
      pout[row * GD + j * 16 + row_in] = acc[j][r];
  }
}

// ---------------------------------------------------------------------------
// reduce: x[0..NPOS) rows = bf16( sum over SPLITK partials )  (deterministic)
// ---------------------------------------------------------------------------
__global__ void reduce_kernel(const float* __restrict__ partial,
                              uint16_t* __restrict__ x) {
  int e = blockIdx.x * blockDim.x + threadIdx.x;    // 0 .. NPOS*GD-1
  float s = 0.0f;
#pragma unroll
  for (int k = 0; k < SPLITK; ++k)
    s += partial[(size_t)k * NPOS * GD + e];
  x[e] = f2bf(s);
}

// ---------------------------------------------------------------------------
// gemm2: y[b, l*128+k] = relu( sum_d x[b,d]*W[l,k,d] + b[l,k] )
// 3 waves / block (one per layer), each wave a 16-row x 128-col tile.
// ---------------------------------------------------------------------------
__global__ void gemm2_kernel(const uint16_t* __restrict__ x,
                             const uint16_t* __restrict__ Wb,
                             const float* __restrict__ bias,
                             float* __restrict__ y) {
  int layer = threadIdx.x >> 5;              // 0..2
  int lane  = threadIdx.x & 31;
  int tile  = blockIdx.x;                    // 0..511 (GB/16)
  int row_in = lane & 15;
  int abase = (lane < 16) ? 0 : 8;
  int koff  = (lane < 16) ? 0 : 8;
  const uint16_t* xr = x + (size_t)(tile * 16 + row_in) * GD;

  v8f acc[8];
#pragma unroll
  for (int j = 0; j < 8; ++j) acc[j] = (v8f){0.f,0.f,0.f,0.f,0.f,0.f,0.f,0.f};

#pragma unroll
  for (int kc = 0; kc < GD; kc += 32) {
    BFrag bf[8];
#pragma unroll
    for (int j = 0; j < 8; ++j) {
      const uint16_t* wr = Wb + ((size_t)layer * GD + j * 16 + row_in) * GD + kc + koff;
      bf[j].u[0] = *(const uint4*)wr;
      bf[j].u[1] = *(const uint4*)(wr + 16);
    }
    BFrag af;
    af.u[0] = *(const uint4*)(xr + kc + abase);
    af.u[1] = *(const uint4*)(xr + kc + abase + 16);
#pragma unroll
    for (int j = 0; j < 8; ++j)
      acc[j] = __builtin_amdgcn_wmma_f32_16x16x32_bf16(
                 false, af.v, false, bf[j].v, (short)0, acc[j], false, false);
  }

  int rowadd = (lane < 16) ? 0 : 8;
#pragma unroll
  for (int j = 0; j < 8; ++j) {
    float bv = bias[layer * GD + j * 16 + row_in];
#pragma unroll
    for (int r = 0; r < 8; ++r) {
      float v = fmaxf(acc[j][r] + bv, 0.0f);
      y[(size_t)(tile * 16 + r + rowadd) * (GL * GD) + layer * GD + j * 16 + row_in] = v;
    }
  }
}

// ---------------------------------------------------------------------------
// launch
// ---------------------------------------------------------------------------
extern "C" void kernel_launch(void* const* d_in, const int* in_sizes, int n_in,
                              void* d_out, int out_size, void* d_ws, size_t ws_size,
                              hipStream_t stream) {
  const int*   labels = (const int*)  d_in[0];
  const int*   nodes  = (const int*)  d_in[1];
  const float* A      = (const float*)d_in[2];
  const float* E      = (const float*)d_in[3];
  const float* W      = (const float*)d_in[4];
  const float* bias   = (const float*)d_in[5];

  float* out        = (float*)d_out;
  float* labels_out = out;            // first GB elements
  float* y          = out + GB;       // GB * GL * GD elements

  // workspace layout (256B aligned)
  char* ws = (char*)d_ws;
  uint16_t* Et      = (uint16_t*)(ws);                               // 4 MiB
  uint16_t* Wb      = (uint16_t*)(ws + 4194304);                     // 96 KiB
  uint16_t* x       = (uint16_t*)(ws + 4194304 + 98304);             // 2 MiB
  int*      pos     = (int*)     (ws + 6389760);                     // 16 KiB
  int*      neg     = (int*)     (ws + 6406144);                     // 16 KiB
  float*    partial = (float*)   (ws + 6422528);                     // 16 MiB

  prep_kernel      <<<(GN * GD) / 256, 256, 0, stream>>>(E, W, Et, Wb);
  partition_kernel <<<1, 1024, 0, stream>>>(labels, nodes, pos, neg, labels_out);
  neg_gather_kernel<<<(NPOS * GD) / 256, 256, 0, stream>>>(E, neg, x);
  gemm1_kernel     <<<(NPOS / 16) * 2, 128, 0, stream>>>(A, Et, pos, partial);
  reduce_kernel    <<<(NPOS * GD) / 256, 256, 0, stream>>>(partial, x);
  gemm2_kernel     <<<GB / 16, 96, 0, stream>>>(x, Wb, bias, y);
}
